// ViTVQ3D_26371099198045
// MI455X (gfx1250) — matrix operand order repaired and checked by reading, not compiled, not run on hardware
//
#include <hip/hip_runtime.h>
#include <math.h>

typedef __bf16 bf16_t;
typedef __attribute__((ext_vector_type(16))) __bf16 v16bf;
typedef __attribute__((ext_vector_type(8)))  float  v8f;

#define EP_BIASPOS 0
#define EP_QKV     1
#define EP_RES     2
#define EP_TANH    3
#define EP_BIAS    4
#define EP_DECONV  5

__device__ __forceinline__ v8f wmma_bf16(v16bf a, v16bf b, v8f c) {
  return __builtin_amdgcn_wmma_f32_16x16x32_bf16(false, a, false, b, (short)0, c, false, false);
}

// ---------------------------------------------------------------------------
// Generic bf16 GEMM, C = A(MxK) * B(KxN), M fixed at 4096, tiles 64x64,
// 8 waves, each wave does one 16x32 output (2 WMMA accumulators).
// K multiple of 32, N multiple of 64 (callers pad with zeros).
// Next-k-tile prefetch via global_prefetch_b8.
// ---------------------------------------------------------------------------
template<int MODE>
__global__ __launch_bounds__(256) void gemm_bf16_kernel(
    const bf16_t* __restrict__ A, int lda,
    const bf16_t* __restrict__ B, int ldb,
    int K,
    float* __restrict__ outf, int ldo,
    const float* __restrict__ bias, int nbias,
    const float* __restrict__ pos,
    bf16_t* __restrict__ outb,
    bf16_t* __restrict__ qp, bf16_t* __restrict__ kp, bf16_t* __restrict__ vp)
{
  __shared__ bf16_t As[64][40];   // [m][k]
  __shared__ bf16_t Bs[64][40];   // [n][k] (B tile stored transposed)

  const int tid  = threadIdx.x;
  const int lane = tid & 31;
  const int w    = tid >> 5;          // 0..7
  const int wm   = w & 3, wn = w >> 2;
  const int m0   = blockIdx.y * 64;
  const int n0   = blockIdx.x * 64;
  const int ml   = lane & 15, g = lane >> 4;

  v8f c0, c1;
  for (int r = 0; r < 8; ++r) { c0[r] = 0.f; c1[r] = 0.f; }

  for (int kk = 0; kk < K; kk += 32) {
    for (int i = tid; i < 64 * 32; i += 256) {
      int r = i >> 5, c = i & 31;
      As[r][c] = A[(size_t)(m0 + r) * lda + (kk + c)];
    }
    for (int i = tid; i < 32 * 64; i += 256) {
      int r = i >> 6, c = i & 63;      // r = k, c = n
      Bs[c][r] = B[(size_t)(kk + r) * ldb + (n0 + c)];
    }
    // prefetch next k-tile while this one is consumed (global_prefetch_b8)
    if (kk + 32 < K) {
      __builtin_prefetch(&A[(size_t)(m0 + (tid >> 2)) * lda + (kk + 32) + (tid & 3) * 8], 0, 3);
      __builtin_prefetch(&B[(size_t)(kk + 32 + (tid >> 3)) * ldb + n0 + (tid & 7) * 8], 0, 3);
    }
    __syncthreads();
    v16bf a, b0, b1;
#pragma unroll
    for (int i = 0; i < 8; ++i) {
      a[i]     = As[wm * 16 + ml][g * 16 + i];
      a[8 + i] = As[wm * 16 + ml][g * 16 + 8 + i];
    }
#pragma unroll
    for (int i = 0; i < 16; ++i) {
      b0[i] = Bs[wn * 32 + ml][g * 16 + i];
      b1[i] = Bs[wn * 32 + 16 + ml][g * 16 + i];
    }
    c0 = wmma_bf16(a, b0, c0);
    c1 = wmma_bf16(a, b1, c1);
    __syncthreads();
  }

  // epilogue
  for (int sub = 0; sub < 2; ++sub) {
    v8f cc = sub ? c1 : c0;
    int nn = n0 + wn * 32 + sub * 16 + ml;
#pragma unroll
    for (int r = 0; r < 8; ++r) {
      int mm = m0 + wm * 16 + r + 8 * g;
      float acc = cc[r];
      if (MODE == EP_BIASPOS) {
        float bb = (bias && nn < nbias) ? bias[nn] : 0.f;
        float pp = pos[(size_t)(mm & 511) * 384 + nn];
        outf[(size_t)mm * ldo + nn] = acc + bb + pp;
      } else if (MODE == EP_RES) {
        float bb = (bias && nn < nbias) ? bias[nn] : 0.f;
        outf[(size_t)mm * ldo + nn] += acc + bb;
      } else if (MODE == EP_TANH) {
        float bb = (bias && nn < nbias) ? bias[nn] : 0.f;
        outb[(size_t)mm * ldo + nn] = (bf16_t)tanhf(acc + bb);
      } else if (MODE == EP_BIAS) {
        float bb = (bias && nn < nbias) ? bias[nn] : 0.f;
        outf[(size_t)mm * ldo + nn] = acc + bb;
      } else if (MODE == EP_QKV) {
        int sec = nn >> 11;                // 0=q 1=k 2=v
        int rr  = nn & 2047;
        int h   = rr >> 7, dd = rr & 127;
        int bi2 = mm >> 9, tok = mm & 511;
        bf16_t* dst = (sec == 0) ? qp : ((sec == 1) ? kp : vp);
        dst[(size_t)((bi2 * 16 + h) * 512 + tok) * 128 + dd] = (bf16_t)acc;
      } else if (MODE == EP_DECONV) {
        float bb = bias[0];
        int bi2 = mm >> 9, tok = mm & 511;
        int d3 = tok >> 6, h3 = (tok >> 3) & 7, w3 = tok & 7;
        int i3 = nn >> 6,  j3 = (nn >> 3) & 7,  k3 = nn & 7;
        size_t off = (size_t)bi2 * 262144 +
                     (size_t)(d3 * 8 + i3) * 4096 + (h3 * 8 + j3) * 64 + (w3 * 8 + k3);
        outf[off] = acc + bb;
      }
    }
  }
}

// ---------------------------------------------------------------------------
// LayerNorm over 360 cols (stride 384), fp32 in -> bf16 out (pad cols zeroed)
// ---------------------------------------------------------------------------
__global__ __launch_bounds__(128) void ln_kernel(const float* __restrict__ x,
                                                 bf16_t* __restrict__ out,
                                                 const float* __restrict__ w,
                                                 const float* __restrict__ b)
{
  __shared__ float red[128];
  const int row = blockIdx.x, t = threadIdx.x;
  const float* xr = x + (size_t)row * 384;
  float s = 0.f;
  for (int c = t; c < 360; c += 128) s += xr[c];
  red[t] = s; __syncthreads();
  for (int st = 64; st > 0; st >>= 1) { if (t < st) red[t] += red[t + st]; __syncthreads(); }
  float mean = red[0] / 360.f;
  __syncthreads();
  float v = 0.f;
  for (int c = t; c < 360; c += 128) { float d = xr[c] - mean; v += d * d; }
  red[t] = v; __syncthreads();
  for (int st = 64; st > 0; st >>= 1) { if (t < st) red[t] += red[t + st]; __syncthreads(); }
  float rstd = rsqrtf(red[0] / 360.f + 1e-5f);
  bf16_t* orow = out + (size_t)row * 384;
  for (int c = t; c < 384; c += 128) {
    float y = 0.f;
    if (c < 360) y = (xr[c] - mean) * rstd * w[c] + b[c];
    orow[c] = (bf16_t)y;
  }
}

// ---------------------------------------------------------------------------
// Fused attention: per (b,h) & 16-query tile. WMMA scores, fp32 softmax,
// WMMA attn@v (v staged through LDS for coalescing), scatter into
// out-proj A layout [4096 x 2048].
// ---------------------------------------------------------------------------
__global__ __launch_bounds__(128) void attn_kernel(const bf16_t* __restrict__ q,
                                                   const bf16_t* __restrict__ k,
                                                   const bf16_t* __restrict__ v,
                                                   bf16_t* __restrict__ o)
{
  __shared__ float  sc[16][512];
  __shared__ bf16_t pb[16][512];
  __shared__ bf16_t vs[32][136];
  __shared__ float  part[16][8];
  __shared__ float  rmax[16];
  __shared__ float  rsum[16];

  const int bh = blockIdx.x >> 5;
  const int q0 = (blockIdx.x & 31) << 4;
  const int tid = threadIdx.x;
  const int lane = tid & 31;
  const int w = tid >> 5;
  const int ml = lane & 15, g = lane >> 4;

  const bf16_t* qp = q + (size_t)bh * 512 * 128;
  const bf16_t* kp = k + (size_t)bh * 512 * 128;
  const bf16_t* vp = v + (size_t)bh * 512 * 128;

  // phase 1: scores = q k^T * rsqrt(128)
  for (int kt = w; kt < 32; kt += 4) {
    v8f acc; for (int r = 0; r < 8; ++r) acc[r] = 0.f;
    for (int kk = 0; kk < 128; kk += 32) {
      v16bf a, b;
#pragma unroll
      for (int i = 0; i < 8; ++i) {
        a[i]     = qp[(size_t)(q0 + ml) * 128 + kk + g * 16 + i];
        a[8 + i] = qp[(size_t)(q0 + ml) * 128 + kk + g * 16 + 8 + i];
      }
#pragma unroll
      for (int i = 0; i < 16; ++i)
        b[i] = kp[(size_t)(kt * 16 + ml) * 128 + kk + g * 16 + i];
      acc = wmma_bf16(a, b, acc);
    }
#pragma unroll
    for (int r = 0; r < 8; ++r)
      sc[r + 8 * g][kt * 16 + ml] = acc[r] * 0.08838834764831845f;
  }
  __syncthreads();

  // phase 2: softmax over 512 keys (row = tid&15, 8 segments of 64)
  const int row = tid & 15, seg = tid >> 4;
  {
    float mx = -3.4e38f;
    for (int c = seg * 64; c < seg * 64 + 64; ++c) mx = fmaxf(mx, sc[row][c]);
    part[row][seg] = mx;
  }
  __syncthreads();
  if (seg == 0) {
    float m2 = part[row][0];
    for (int i = 1; i < 8; ++i) m2 = fmaxf(m2, part[row][i]);
    rmax[row] = m2;
  }
  __syncthreads();
  {
    float mrow = rmax[row];
    float s = 0.f;
    for (int c = seg * 64; c < seg * 64 + 64; ++c) {
      float e = __expf(sc[row][c] - mrow);
      sc[row][c] = e; s += e;
    }
    part[row][seg] = s;
  }
  __syncthreads();
  if (seg == 0) {
    float s2 = 0.f;
    for (int i = 0; i < 8; ++i) s2 += part[row][i];
    rsum[row] = s2;
  }
  __syncthreads();
  {
    float inv = 1.f / rsum[row];
    for (int c = seg * 64; c < seg * 64 + 64; ++c)
      pb[row][c] = (bf16_t)(sc[row][c] * inv);
  }
  __syncthreads();

  // phase 3: out = probs @ v. Stage each 32x128 v-slab in LDS (coalesced),
  // each wave accumulates its two 16-wide d-tiles.
  const int bidx = bh >> 4, h = bh & 15;
  const int dt0 = w * 2, dt1 = dt0 + 1;
  v8f acc0, acc1;
  for (int r = 0; r < 8; ++r) { acc0[r] = 0.f; acc1[r] = 0.f; }
  for (int kt = 0; kt < 16; ++kt) {
    for (int i = tid; i < 32 * 128; i += 128) {
      int r = i >> 7, c = i & 127;
      vs[r][c] = vp[(size_t)(kt * 32 + r) * 128 + c];
    }
    __syncthreads();
    v16bf a, b0, b1;
#pragma unroll
    for (int i = 0; i < 8; ++i) {
      a[i]     = pb[ml][kt * 32 + g * 16 + i];
      a[8 + i] = pb[ml][kt * 32 + g * 16 + 8 + i];
    }
#pragma unroll
    for (int i = 0; i < 16; ++i) {
      b0[i] = vs[g * 16 + i][dt0 * 16 + ml];
      b1[i] = vs[g * 16 + i][dt1 * 16 + ml];
    }
    acc0 = wmma_bf16(a, b0, acc0);
    acc1 = wmma_bf16(a, b1, acc1);
    __syncthreads();
  }
#pragma unroll
  for (int r = 0; r < 8; ++r) {
    size_t base = (size_t)(bidx * 512 + q0 + r + 8 * g) * 2048 + h * 128;
    o[base + dt0 * 16 + ml] = (bf16_t)acc0[r];
    o[base + dt1 * 16 + ml] = (bf16_t)acc1[r];
  }
}

// ---------------------------------------------------------------------------
// fp32 -> bf16 weight convert with zero padding (optional transpose)
// dst[k*Np+n] = (k<sK && n<sN) ? (tr ? src[n*sK+k] : src[k*sN+n]) : 0
// ---------------------------------------------------------------------------
__global__ __launch_bounds__(256) void cvt_pad_kernel(const float* __restrict__ src,
                                                      bf16_t* __restrict__ dst,
                                                      int Kp, int Np, int sK, int sN, int tr)
{
  int total = Kp * Np;
  int i = blockIdx.x * 256 + threadIdx.x;
  if (i >= total) return;
  int kk = i / Np, nn = i - kk * Np;
  float v = 0.f;
  if (kk < sK && nn < sN)
    v = tr ? src[(size_t)nn * sK + kk] : src[(size_t)kk * sN + nn];
  dst[i] = (bf16_t)v;
}

// ---------------------------------------------------------------------------
// 3D sincos positional embedding: pos[512][384], pad cols zero
// ---------------------------------------------------------------------------
__global__ void pos_kernel(float* __restrict__ pos)
{
  int n = blockIdx.x;       // token 0..511  (n = a*64 + b*8 + c)
  int j = threadIdx.x;      // col 0..383
  float val = 0.f;
  if (j < 360) {
    int comp = j / 120, jj = j - comp * 120;
    int half = jj / 60, d = jj - half * 60;
    float omega = powf(10000.f, -((float)d / 60.f));
    int a = n >> 6, bb = (n >> 3) & 7, c = n & 7;
    float p = (comp == 0) ? (float)bb : ((comp == 1) ? (float)a : (float)c);
    float x = p * omega;
    val = half ? __cosf(x) : __sinf(x);
  }
  pos[(size_t)n * 384 + j] = val;
}

// ---------------------------------------------------------------------------
// patch extraction: volume (B,1,64,64,64) -> A bf16 [4096][512]
// ---------------------------------------------------------------------------
__global__ __launch_bounds__(256) void patch_kernel(const float* __restrict__ vol,
                                                    bf16_t* __restrict__ dst)
{
  int m = blockIdx.x, t = threadIdx.x;
  int b = m >> 9, tok = m & 511;
  int d = tok >> 6, h = (tok >> 3) & 7, w = tok & 7;
  const float* vb = vol + (size_t)b * 262144;
  for (int c = t; c < 512; c += 256) {
    int i = c >> 6, j = (c >> 3) & 7, kk = c & 7;
    dst[(size_t)m * 512 + c] =
        (bf16_t)vb[(size_t)(d * 8 + i) * 4096 + (h * 8 + j) * 64 + (w * 8 + kk)];
  }
}

// ---------------------------------------------------------------------------
// Vector quantization: one block per token row. Euclidean argmin over 1024
// codes, write selected code (bf16) and per-row commit loss (deterministic).
// ---------------------------------------------------------------------------
__global__ __launch_bounds__(256) void vq_kernel(const float* __restrict__ z,
                                                 const float* __restrict__ cb,
                                                 int* __restrict__ idx,
                                                 bf16_t* __restrict__ qc,
                                                 float* __restrict__ rowloss)
{
  __shared__ float zr[256];
  __shared__ float bd[256];
  __shared__ int   bi[256];
  const int row = blockIdx.x, t = threadIdx.x;
  zr[t] = z[(size_t)row * 256 + t];
  __syncthreads();
  float best = 3.4e38f; int bn = 0;
  for (int n = t; n < 1024; n += 256) {
    const float* c = cb + (size_t)n * 256;
    float s = 0.f;
    for (int kk = 0; kk < 256; ++kk) { float d = zr[kk] - c[kk]; s = fmaf(d, d, s); }
    if (s < best) { best = s; bn = n; }
  }
  bd[t] = best; bi[t] = bn;
  __syncthreads();
  for (int st = 128; st > 0; st >>= 1) {
    if (t < st) {
      if (bd[t + st] < bd[t] || (bd[t + st] == bd[t] && bi[t + st] < bi[t])) {
        bd[t] = bd[t + st]; bi[t] = bi[t + st];
      }
    }
    __syncthreads();
  }
  int win = bi[0];
  if (t == 0) idx[row] = win;
  float cv = cb[(size_t)win * 256 + t];
  qc[(size_t)row * 256 + t] = (bf16_t)cv;
  float d = cv - zr[t];
  __syncthreads();
  bd[t] = d * d;
  __syncthreads();
  for (int st = 128; st > 0; st >>= 1) { if (t < st) bd[t] += bd[t + st]; __syncthreads(); }
  if (t == 0) rowloss[row] = bd[0];
}

__global__ __launch_bounds__(256) void loss_reduce_kernel(const float* __restrict__ rl,
                                                          float* __restrict__ out)
{
  __shared__ float red[256];
  int t = threadIdx.x;
  float s = 0.f;
  for (int i = t; i < 4096; i += 256) s += rl[i];
  red[t] = s; __syncthreads();
  for (int st = 128; st > 0; st >>= 1) { if (t < st) red[t] += red[t + st]; __syncthreads(); }
  if (t == 0) out[0] = red[0] * (1.f / (4096.f * 256.f));
}

// ---------------------------------------------------------------------------
// host-side transformer runner (6 layers)
// ---------------------------------------------------------------------------
static void run_tf(hipStream_t stream, float* x, bf16_t* xb, bf16_t* wbuf,
                   bf16_t* qbuf, bf16_t* kbuf, bf16_t* vbuf, bf16_t* obuf,
                   const float* ln1w, const float* ln1b, const float* qkvw,
                   const float* outw, const float* outb, const float* ln2w,
                   const float* ln2b, const float* ff1w, const float* ff1b,
                   const float* ff2w, const float* ff2b)
{
  for (int l = 0; l < 6; ++l) {
    ln_kernel<<<4096, 128, 0, stream>>>(x, xb, ln1w + l * 360, ln1b + l * 360);

    cvt_pad_kernel<<<(384 * 6144 + 255) / 256, 256, 0, stream>>>(
        qkvw + (size_t)l * 360 * 6144, wbuf, 384, 6144, 360, 6144, 0);
    gemm_bf16_kernel<EP_QKV><<<dim3(96, 64), 256, 0, stream>>>(
        xb, 384, wbuf, 6144, 384, nullptr, 0, nullptr, 0, nullptr, nullptr,
        qbuf, kbuf, vbuf);

    attn_kernel<<<4096, 128, 0, stream>>>(qbuf, kbuf, vbuf, obuf);

    cvt_pad_kernel<<<(2048 * 384 + 255) / 256, 256, 0, stream>>>(
        outw + (size_t)l * 2048 * 360, wbuf, 2048, 384, 2048, 360, 0);
    gemm_bf16_kernel<EP_RES><<<dim3(6, 64), 256, 0, stream>>>(
        obuf, 2048, wbuf, 384, 2048, x, 384, outb + l * 360, 360, nullptr,
        nullptr, nullptr, nullptr, nullptr);

    ln_kernel<<<4096, 128, 0, stream>>>(x, xb, ln2w + l * 360, ln2b + l * 360);

    cvt_pad_kernel<<<(384 * 1024 + 255) / 256, 256, 0, stream>>>(
        ff1w + (size_t)l * 360 * 1024, wbuf, 384, 1024, 360, 1024, 0);
    gemm_bf16_kernel<EP_TANH><<<dim3(16, 64), 256, 0, stream>>>(
        xb, 384, wbuf, 1024, 384, nullptr, 1024, ff1b + l * 1024, 1024, nullptr,
        obuf, nullptr, nullptr, nullptr);

    cvt_pad_kernel<<<(1024 * 384 + 255) / 256, 256, 0, stream>>>(
        ff2w + (size_t)l * 1024 * 360, wbuf, 1024, 384, 1024, 360, 0);
    gemm_bf16_kernel<EP_RES><<<dim3(6, 64), 256, 0, stream>>>(
        obuf, 1024, wbuf, 384, 1024, x, 384, ff2b + l * 360, 360, nullptr,
        nullptr, nullptr, nullptr, nullptr);
  }
}

extern "C" void kernel_launch(void* const* d_in, const int* in_sizes, int n_in,
                              void* d_out, int out_size, void* d_ws, size_t ws_size,
                              hipStream_t stream)
{
  (void)in_sizes; (void)n_in; (void)out_size; (void)ws_size;

  // inputs (setup_inputs dict order, nested dicts flattened depth-first)
  const float* volume   = (const float*)d_in[0];
  const float* conv_w   = (const float*)d_in[1];
  const float* conv_b   = (const float*)d_in[2];
  const float* e_ln1w   = (const float*)d_in[3];
  const float* e_ln1b   = (const float*)d_in[4];
  const float* e_qkvw   = (const float*)d_in[5];
  const float* e_outw   = (const float*)d_in[6];
  const float* e_outb   = (const float*)d_in[7];
  const float* e_ln2w   = (const float*)d_in[8];
  const float* e_ln2b   = (const float*)d_in[9];
  const float* e_ff1w   = (const float*)d_in[10];
  const float* e_ff1b   = (const float*)d_in[11];
  const float* e_ff2w   = (const float*)d_in[12];
  const float* e_ff2b   = (const float*)d_in[13];
  const float* e_lnw    = (const float*)d_in[14];
  const float* e_lnb    = (const float*)d_in[15];
  const float* pre_w    = (const float*)d_in[16];
  const float* pre_b    = (const float*)d_in[17];
  const float* codebook = (const float*)d_in[18];
  const float* post_w   = (const float*)d_in[19];
  const float* post_b   = (const float*)d_in[20];
  const float* d_ln1w   = (const float*)d_in[21];
  const float* d_ln1b   = (const float*)d_in[22];
  const float* d_qkvw   = (const float*)d_in[23];
  const float* d_outw   = (const float*)d_in[24];
  const float* d_outb   = (const float*)d_in[25];
  const float* d_ln2w   = (const float*)d_in[26];
  const float* d_ln2b   = (const float*)d_in[27];
  const float* d_ff1w   = (const float*)d_in[28];
  const float* d_ff1b   = (const float*)d_in[29];
  const float* d_ff2w   = (const float*)d_in[30];
  const float* d_ff2b   = (const float*)d_in[31];
  const float* d_lnw    = (const float*)d_in[32];
  const float* d_lnb    = (const float*)d_in[33];
  const float* deconv_w = (const float*)d_in[34];
  const float* deconv_b = (const float*)d_in[35];

  float* rec    = (float*)d_out;               // 8*64^3 = 2,097,152 floats
  float* d_loss = rec + 2097152;               // commit loss scalar

  // workspace carve-up (~90 MB)
  char* p = (char*)d_ws;
  auto carve = [&](size_t bytes) {
    void* r = (void*)p;
    p += (bytes + 255) & ~(size_t)255;
    return r;
  };
  float*  x       = (float*) carve(4096ull * 384 * 4);
  bf16_t* xb      = (bf16_t*)carve(4096ull * 384 * 2);
  bf16_t* wbuf    = (bf16_t*)carve(384ull * 6144 * 2);
  bf16_t* qbuf    = (bf16_t*)carve(4096ull * 2048 * 2);
  bf16_t* kbuf    = (bf16_t*)carve(4096ull * 2048 * 2);
  bf16_t* vbuf    = (bf16_t*)carve(4096ull * 2048 * 2);
  bf16_t* obuf    = (bf16_t*)carve(4096ull * 2048 * 2);  // also patches / mlp hidden
  float*  z       = (float*) carve(4096ull * 256 * 4);
  bf16_t* qcode   = (bf16_t*)carve(4096ull * 256 * 2);
  int*    idxb    = (int*)   carve(4096ull * 4);
  float*  posb    = (float*) carve(512ull * 384 * 4);
  float*  rowloss = (float*) carve(4096ull * 4);

  // positional embedding (module constant in the reference)
  pos_kernel<<<512, 384, 0, stream>>>(posb);

  // ---- encoder: patch conv (as GEMM) + pos ----
  patch_kernel<<<4096, 256, 0, stream>>>(volume, obuf);
  cvt_pad_kernel<<<(512 * 384 + 255) / 256, 256, 0, stream>>>(
      conv_w, wbuf, 512, 384, 512, 360, 1);   // transpose (DIM,512)->(512,384)
  gemm_bf16_kernel<EP_BIASPOS><<<dim3(6, 64), 256, 0, stream>>>(
      obuf, 512, wbuf, 384, 512, x, 384, conv_b, 360, posb,
      nullptr, nullptr, nullptr, nullptr);

  run_tf(stream, x, xb, wbuf, qbuf, kbuf, vbuf, obuf,
         e_ln1w, e_ln1b, e_qkvw, e_outw, e_outb, e_ln2w, e_ln2b,
         e_ff1w, e_ff1b, e_ff2w, e_ff2b);
  ln_kernel<<<4096, 128, 0, stream>>>(x, xb, e_lnw, e_lnb);

  // ---- pre-quant projection ----
  cvt_pad_kernel<<<(384 * 256 + 255) / 256, 256, 0, stream>>>(
      pre_w, wbuf, 384, 256, 360, 256, 0);
  gemm_bf16_kernel<EP_BIAS><<<dim3(4, 64), 256, 0, stream>>>(
      xb, 384, wbuf, 256, 384, z, 256, pre_b, 256, nullptr,
      nullptr, nullptr, nullptr, nullptr);

  // ---- vector quantization + commit loss ----
  vq_kernel<<<4096, 256, 0, stream>>>(z, codebook, idxb, qcode, rowloss);
  loss_reduce_kernel<<<1, 256, 0, stream>>>(rowloss, d_loss);

  // ---- post-quant projection + pos ----
  cvt_pad_kernel<<<(256 * 384 + 255) / 256, 256, 0, stream>>>(
      post_w, wbuf, 256, 384, 256, 360, 0);
  gemm_bf16_kernel<EP_BIASPOS><<<dim3(6, 64), 256, 0, stream>>>(
      qcode, 256, wbuf, 384, 256, x, 384, post_b, 360, posb,
      nullptr, nullptr, nullptr, nullptr);

  // ---- decoder ----
  run_tf(stream, x, xb, wbuf, qbuf, kbuf, vbuf, obuf,
         d_ln1w, d_ln1b, d_qkvw, d_outw, d_outb, d_ln2w, d_ln2b,
         d_ff1w, d_ff1b, d_ff2w, d_ff2b);
  ln_kernel<<<4096, 128, 0, stream>>>(x, xb, d_lnw, d_lnb);

  // ---- deconv (as GEMM, scatter into volume) ----
  cvt_pad_kernel<<<(384 * 512 + 255) / 256, 256, 0, stream>>>(
      deconv_w, wbuf, 384, 512, 360, 512, 0);
  gemm_bf16_kernel<EP_DECONV><<<dim3(8, 64), 256, 0, stream>>>(
      xb, 384, wbuf, 512, 384, rec, 0, deconv_b, 1, nullptr,
      nullptr, nullptr, nullptr, nullptr);
}
